// Perceiver_2310692405281
// MI455X (gfx1250) — matrix operand order repaired and checked by reading, not compiled, not run on hardware
//
#include <hip/hip_runtime.h>
#include <hip/hip_bf16.h>
#include <cstdint>

typedef __attribute__((ext_vector_type(16))) _Float16 v16h;
typedef __attribute__((ext_vector_type(8)))  float    v8f;

// ---------------- problem constants ----------------
#define BATCH   4096
#define TT      96
#define FRAW    5
#define TOTF    11
#define HID     512
#define LAYERS  4
#define HEADS   8
#define DH      64
#define PATCH   8
#define NPATCH  12
#define PDIM    88      // padded to 128 for K%64==0
#define PDIMP   128
#define ROWS    (BATCH*NPATCH)   // 49152
#define MLPH    2048
#define CLSK    (HID*NPATCH)     // 6144

// GEMM epilogue flags
#define FBIAS  1
#define FGELU  2
#define FRESID 4
#define FF32   8
#define FF16   16

// =====================================================================
// f16 WMMA GEMM: C(MxN) = A(MxK) * Bt(NxK)^T  [+bias][gelu][resid]
// A row-major MxK, B supplied TRANSPOSED (N x K, K-contiguous rows) so
// both operand tiles are plain row-major copies. Tile 128x128x64,
// 256 threads (8 waves), wave tile 64x32, 16 WMMA / K-step.
// Both tiles staged with GLOBAL_LOAD_ASYNC_TO_LDS_B128 (ASYNCcnt),
// double-buffered in LDS (64 KB).
// =====================================================================
template<int FLAGS>
__global__ __launch_bounds__(256)
void gemm_f16(const _Float16* __restrict__ A, const _Float16* __restrict__ Bt,
              const float* __restrict__ bias, float* __restrict__ Cf32,
              _Float16* __restrict__ Cf16, int M, int N, int K)
{
    // [buf][row][64 halfs]; buf stride 8192 halfs = 16384 B
    __shared__ __align__(16) _Float16 As[2 * 128 * 64];
    __shared__ __align__(16) _Float16 Bs[2 * 128 * 64];

    const int tid  = threadIdx.x;
    const int m0   = blockIdx.y * 128;
    const int n0   = blockIdx.x * 128;
    const int wave = tid >> 5;
    const int lane = tid & 31;
    const int wm   = (wave >> 2) * 64;   // wave M offset within tile
    const int wn   = (wave & 3) * 32;    // wave N offset within tile
    const int lhi  = lane >> 4;          // 0/1 half-wave
    const int llo  = lane & 15;

    v8f acc[4][2];
    #pragma unroll
    for (int mt = 0; mt < 4; ++mt)
        #pragma unroll
        for (int nt = 0; nt < 2; ++nt)
            acc[mt][nt] = v8f{0.f,0.f,0.f,0.f,0.f,0.f,0.f,0.f};

    // per-thread staging addresses: row = (tid>>3)+32*it, col chunk = (tid&7)*8
    const int srow = tid >> 3;
    const int scol = (tid & 7) << 3;
    const _Float16* gA = A  + (size_t)(m0 + srow) * K + scol;
    const _Float16* gB = Bt + (size_t)(n0 + srow) * K + scol;
    const unsigned ldsA0 = (unsigned)(uintptr_t)(&As[srow * 64 + scol]);
    const unsigned ldsB0 = (unsigned)(uintptr_t)(&Bs[srow * 64 + scol]);

    // copy 128x64 halfs of A and Bt into LDS buffer (async, 16B/lane/op)
    auto stage = [&](int buf) {
        const unsigned bofs = (unsigned)buf * 16384u;
        #pragma unroll
        for (int it = 0; it < 4; ++it) {
            unsigned la = ldsA0 + bofs + (unsigned)it * (32u * 128u);
            unsigned lb = ldsB0 + bofs + (unsigned)it * (32u * 128u);
            const _Float16* pa = gA + (size_t)it * 32 * K;
            const _Float16* pb = gB + (size_t)it * 32 * K;
            asm volatile("global_load_async_to_lds_b128 %0, %1, off"
                         :: "v"(la), "v"(pa) : "memory");
            asm volatile("global_load_async_to_lds_b128 %0, %1, off"
                         :: "v"(lb), "v"(pb) : "memory");
        }
        gA += 64; gB += 64;
    };

    // fragments + 16 WMMAs from one LDS buffer (two 32-wide K chunks)
    auto compute = [&](int buf) {
        const int bofs = buf * 8192;
        #pragma unroll
        for (int kc = 0; kc < 64; kc += 32) {
            v16h af[4], bf[2];
            #pragma unroll
            for (int mt = 0; mt < 4; ++mt) {
                const _Float16* pa = &As[bofs + (wm + mt * 16 + llo) * 64 + kc + lhi * 8];
                union { v16h v; float4 f[2]; } u;
                u.f[0] = *(const float4*)(pa);        // K = kc+kb .. +7
                u.f[1] = *(const float4*)(pa + 16);   // K = kc+16+kb .. +7
                af[mt] = u.v;
            }
            #pragma unroll
            for (int nt = 0; nt < 2; ++nt) {
                const _Float16* pb = &Bs[bofs + (wn + nt * 16 + llo) * 64 + kc + lhi * 8];
                union { v16h v; float4 f[2]; } u;
                u.f[0] = *(const float4*)(pb);        // K = kc+kb .. +7
                u.f[1] = *(const float4*)(pb + 16);   // K = kc+16+kb .. +7
                bf[nt] = u.v;
            }
            #pragma unroll
            for (int mt = 0; mt < 4; ++mt)
                #pragma unroll
                for (int nt = 0; nt < 2; ++nt)
                    acc[mt][nt] = __builtin_amdgcn_wmma_f32_16x16x32_f16(
                        false, af[mt], false, bf[nt], (short)0, acc[mt][nt],
                        false, false);
        }
    };

    // ---- pipelined main loop ----
    stage(0);
    asm volatile("s_wait_asynccnt 0x0" ::: "memory");
    __syncthreads();

    const int nk = K >> 6;
    for (int kk = 0; kk < nk; ++kk) {
        const int cur = kk & 1;
        if (kk + 1 < nk) stage(cur ^ 1);      // async copy overlaps compute
        compute(cur);
        if (kk + 1 < nk)
            asm volatile("s_wait_asynccnt 0x0" ::: "memory");
        __syncthreads();
    }

    // ---- epilogue ----
    #pragma unroll
    for (int mt = 0; mt < 4; ++mt) {
        #pragma unroll
        for (int nt = 0; nt < 2; ++nt) {
            #pragma unroll
            for (int r = 0; r < 8; ++r) {
                int row = m0 + wm + mt * 16 + r + lhi * 8;
                int col = n0 + wn + nt * 16 + llo;
                float v = acc[mt][nt][r];
                if constexpr (FLAGS & FBIAS) v += bias[col];
                if constexpr (FLAGS & FGELU)
                    v = 0.5f * v * (1.0f + erff(v * 0.70710678118654752f));
                size_t o = (size_t)row * N + col;
                if constexpr (FLAGS & FRESID) Cf32[o] += v;
                else if constexpr (FLAGS & FF32) Cf32[o] = v;
                if constexpr (FLAGS & FF16) Cf16[o] = (_Float16)v;
            }
        }
    }
}

// =====================================================================
// Technical factors: 1 thread per batch row, row staged in LDS.
// feats (B,96,11) fp32 = [x(5) | macd,rsi,bb,log_ret,lvr,vmr]
// =====================================================================
__global__ __launch_bounds__(64)
void factors_kernel(const float* __restrict__ x, float* __restrict__ feats)
{
    __shared__ float cs[64 * 96];
    __shared__ float vs[64 * 96];
    const int tid = threadIdx.x;
    const int b   = blockIdx.x * 64 + tid;
    const float* xb = x + (size_t)b * TT * FRAW;
    float* cr = &cs[tid * 96];
    float* vr = &vs[tid * 96];
    for (int t = 0; t < TT; ++t) {
        cr[t] = xb[t * FRAW + 3];
        float v = xb[t * FRAW + 4];
        vr[t] = v > 0.f ? v : 1.f;          // safe_vol
    }
    const float a13 = 2.f / 13.f, a27 = 2.f / 27.f, a14 = 1.f / 14.f;
    float e13 = cr[0], e27 = cr[0], ag = 0.f, al = 0.f;
    float sc = 0.f, sc2 = 0.f, sv = 0.f;
    float* fb = feats + (size_t)b * TT * TOTF;
    for (int t = 0; t < TT; ++t) {
        float c = cr[t];
        if (t > 0) {
            e13 = a13 * c + (1.f - a13) * e13;
            e27 = a27 * c + (1.f - a27) * e27;
            float d = c - cr[t - 1];
            float g = d > 0.f ? d : 0.f;
            float l = d < 0.f ? -d : 0.f;
            ag = a14 * g + (1.f - a14) * ag;
            al = a14 * l + (1.f - a14) * al;
        }
        float macd = e13 - e27;
        float rs   = ag / (al + 1e-10f);
        float rsi  = (100.f - 100.f / (1.f + rs)) * 0.01f;
        sc += c; sc2 += c * c;
        float svol = vr[t];
        sv += svol;
        if (t >= 20) {
            float co = cr[t - 20];
            sc -= co; sc2 -= co * co; sv -= vr[t - 20];
        }
        float bb = 0.f, vmr = 1.f;
        if (t >= 19) {
            float mean = sc * 0.05f;
            float var  = sc2 * 0.05f - mean * mean;
            float sd   = sqrtf(var > 0.f ? var : 0.f);
            bb  = 4.f * sd / (mean + 1e-8f);
            vmr = svol / (sv * 0.05f + 1e-8f);
        }
        float lr = 0.f, lv = 0.f;
        if (t > 0) {
            float pc = cr[t - 1] > 1e-8f ? cr[t - 1] : 1e-8f;
            float cc = c > 1e-8f ? c : 1e-8f;
            lr = logf(cc / pc);
            lv = logf(svol / (vr[t - 1] + 1e-8f));
        }
        float fac[6] = {macd, rsi, bb, lr, lv, vmr};
        float* o = fb + t * TOTF;
        #pragma unroll
        for (int j = 0; j < FRAW; ++j) o[j] = xb[t * FRAW + j];
        #pragma unroll
        for (int j = 0; j < 6; ++j) {
            float f = fac[j];
            o[FRAW + j] = __builtin_isfinite(f) ? f : 0.f;
        }
    }
}

// =====================================================================
// Per-(b,feature) time-normalization + patch pack into f16 A0 (ROWS x 128)
// A0[b*12+p][j*11+f] = normed feats[b, 8p+j, f]; cols 88..127 zero-padded
// =====================================================================
__global__ void normpack_kernel(const float* __restrict__ feats,
                                const float* __restrict__ in_w,
                                const float* __restrict__ in_b,
                                _Float16* __restrict__ A0)
{
    int idx = blockIdx.x * blockDim.x + threadIdx.x;
    if (idx >= BATCH * TOTF) return;
    int b = idx / TOTF, f = idx % TOTF;
    const float* p = feats + (size_t)b * TT * TOTF + f;
    float s = 0.f, s2 = 0.f;
    for (int t = 0; t < TT; ++t) { float v = p[t * TOTF]; s += v; s2 += v * v; }
    float m   = s * (1.f / TT);
    float var = s2 * (1.f / TT) - m * m;
    float inv = rsqrtf(var + 1e-5f);
    float w = in_w[f], bb = in_b[f];
    _Float16* ab = A0 + (size_t)b * NPATCH * PDIMP;
    for (int t = 0; t < TT; ++t) {
        float v = (p[t * TOTF] - m) * inv * w + bb;
        ab[(t >> 3) * PDIMP + (t & 7) * TOTF + f] = (_Float16)v;
    }
    if (f == 0) {
        for (int pp = 0; pp < NPATCH; ++pp)
            for (int c = PDIM; c < PDIMP; ++c)
                ab[pp * PDIMP + c] = (_Float16)0.f;
    }
}

// =====================================================================
// LayerNorm: one wave32 per 512-wide row, fp32 in -> f16 out
// =====================================================================
__global__ __launch_bounds__(256)
void ln_kernel(const float* __restrict__ h, const float* __restrict__ w,
               const float* __restrict__ b, _Float16* __restrict__ out)
{
    int row  = blockIdx.x * 8 + (threadIdx.x >> 5);
    int lane = threadIdx.x & 31;
    const float* x = h + (size_t)row * HID;
    float s = 0.f, s2 = 0.f;
    for (int i = lane; i < HID; i += 32) { float v = x[i]; s += v; s2 += v * v; }
    #pragma unroll
    for (int off = 16; off > 0; off >>= 1) {
        s  += __shfl_xor(s,  off, 32);
        s2 += __shfl_xor(s2, off, 32);
    }
    float m   = s * (1.f / HID);
    float var = s2 * (1.f / HID) - m * m;
    float inv = rsqrtf(var + 1e-5f);
    _Float16* o = out + (size_t)row * HID;
    for (int i = lane; i < HID; i += 32)
        o[i] = (_Float16)((x[i] - m) * inv * w[i] + b[i]);
}

// =====================================================================
// Attention over 12 tokens: one block per batch elem, thread=(head,q)
// qkv f16 (ROWS x 1536) -> o f16 (ROWS x 512), layout [q][h*64+d]
// =====================================================================
__global__ __launch_bounds__(96)
void attn_kernel(const _Float16* __restrict__ qkv, _Float16* __restrict__ o)
{
    __shared__ float ks[NPATCH * HID];
    __shared__ float vsd[NPATCH * HID];
    const int b   = blockIdx.x;
    const int tid = threadIdx.x;
    for (int i = tid; i < NPATCH * HID; i += 96) {
        int t = i / HID, c = i % HID;
        size_t base = (size_t)(b * NPATCH + t) * (3 * HID);
        ks[i]  = (float)qkv[base + HID + c];
        vsd[i] = (float)qkv[base + 2 * HID + c];
    }
    __syncthreads();
    const int h  = tid / NPATCH;
    const int qi = tid % NPATCH;
    float qv[DH];
    size_t qbase = (size_t)(b * NPATCH + qi) * (3 * HID) + h * DH;
    #pragma unroll
    for (int d = 0; d < DH; ++d) qv[d] = (float)qkv[qbase + d];
    float sc[NPATCH];
    float mx = -3.4e38f;
    #pragma unroll
    for (int k = 0; k < NPATCH; ++k) {
        float s = 0.f;
        #pragma unroll
        for (int d = 0; d < DH; ++d) s += qv[d] * ks[k * HID + h * DH + d];
        s *= 0.125f;                 // 1/sqrt(64)
        sc[k] = s;
        mx = s > mx ? s : mx;
    }
    float den = 0.f;
    #pragma unroll
    for (int k = 0; k < NPATCH; ++k) { sc[k] = expf(sc[k] - mx); den += sc[k]; }
    float inv = 1.f / den;
    _Float16* ob = o + (size_t)(b * NPATCH + qi) * HID + h * DH;
    #pragma unroll
    for (int d = 0; d < DH; ++d) {
        float acc = 0.f;
        #pragma unroll
        for (int k = 0; k < NPATCH; ++k) acc += sc[k] * vsd[k * HID + h * DH + d];
        ob[d] = (_Float16)(acc * inv);
    }
}

// =====================================================================
// weight staging: f32 (KxN) -> f16 TRANSPOSED (NxK); plus pe pad variant
// =====================================================================
__global__ void castT_kernel(const float* __restrict__ src,
                             _Float16* __restrict__ dst, int K, int N)
{
    int idx = blockIdx.x * blockDim.x + threadIdx.x;
    if (idx >= K * N) return;
    int k = idx / N, n = idx % N;
    dst[(size_t)n * K + k] = (_Float16)src[idx];
}

__global__ void peT_kernel(const float* __restrict__ src,
                           _Float16* __restrict__ dst)
{
    int idx = blockIdx.x * blockDim.x + threadIdx.x;   // over HID*PDIMP (NxK)
    if (idx >= HID * PDIMP) return;
    int n = idx / PDIMP, k = idx % PDIMP;
    dst[idx] = (k < PDIM) ? (_Float16)src[(size_t)k * HID + n] : (_Float16)0.f;
}

__global__ void cast_kernel(const float* __restrict__ src,
                            _Float16* __restrict__ dst, int n)
{
    int i = blockIdx.x * blockDim.x + threadIdx.x;
    if (i < n) dst[i] = (_Float16)src[i];
}

__global__ void cls3_kernel(const _Float16* __restrict__ f2,
                            const float* __restrict__ w,
                            const float* __restrict__ b,
                            float* __restrict__ out)
{
    int bi = blockIdx.x * blockDim.x + threadIdx.x;
    if (bi >= BATCH) return;
    float a0 = b[0], a1 = b[1], a2 = b[2];
    const _Float16* r = f2 + (size_t)bi * 128;
    for (int k = 0; k < 128; ++k) {
        float v = (float)r[k];
        a0 += v * w[k * 3 + 0];
        a1 += v * w[k * 3 + 1];
        a2 += v * w[k * 3 + 2];
    }
    out[bi * 3 + 0] = a0; out[bi * 3 + 1] = a1; out[bi * 3 + 2] = a2;
}

// =====================================================================
// host launcher
// =====================================================================
extern "C" void kernel_launch(void* const* d_in, const int* in_sizes, int n_in,
                              void* d_out, int out_size, void* d_ws, size_t ws_size,
                              hipStream_t stream)
{
    const float* x      = (const float*)d_in[0];
    const float* in_w   = (const float*)d_in[1];
    const float* in_b   = (const float*)d_in[2];
    const float* pe_w   = (const float*)d_in[3];
    const float* pe_b   = (const float*)d_in[4];
    const float* ln1_w  = (const float*)d_in[5];
    const float* ln1_b  = (const float*)d_in[6];
    const float* qkv_w  = (const float*)d_in[7];
    const float* qkv_b  = (const float*)d_in[8];
    const float* out_w  = (const float*)d_in[9];
    const float* out_b  = (const float*)d_in[10];
    const float* ln2_w  = (const float*)d_in[11];
    const float* ln2_b  = (const float*)d_in[12];
    const float* mlp_w1 = (const float*)d_in[13];
    const float* mlp_b1 = (const float*)d_in[14];
    const float* mlp_w2 = (const float*)d_in[15];
    const float* mlp_b2 = (const float*)d_in[16];
    const float* cls_w1 = (const float*)d_in[17];
    const float* cls_b1 = (const float*)d_in[18];
    const float* cls_w2 = (const float*)d_in[19];
    const float* cls_b2 = (const float*)d_in[20];
    const float* cls_w3 = (const float*)d_in[21];
    const float* cls_b3 = (const float*)d_in[22];

    char* ws = (char*)d_ws;

    // ---- workspace layout (bytes); all weights stored transposed ----
    const size_t W_PE  = 0;                                   // 512x128 f16
    const size_t W_QKV = W_PE  + (size_t)HID * PDIMP * 2;
    const size_t W_OUT = W_QKV + (size_t)LAYERS * HID * 3 * HID * 2;
    const size_t W_M1  = W_OUT + (size_t)LAYERS * HID * HID * 2;
    const size_t W_M2  = W_M1  + (size_t)LAYERS * HID * MLPH * 2;
    const size_t W_C1  = W_M2  + (size_t)LAYERS * MLPH * HID * 2;
    const size_t W_C2  = W_C1  + (size_t)CLSK * HID * 2;
    const size_t H_OFF = W_C2  + (size_t)HID * 128 * 2;
    const size_t S1    = H_OFF + (size_t)ROWS * HID * 4;      // 151 MB region
    const size_t S2    = S1    + (size_t)ROWS * 3 * HID * 2;  // 201 MB region

    _Float16* wpe  = (_Float16*)(ws + W_PE);
    _Float16* wqkv = (_Float16*)(ws + W_QKV);
    _Float16* wout = (_Float16*)(ws + W_OUT);
    _Float16* wm1  = (_Float16*)(ws + W_M1);
    _Float16* wm2  = (_Float16*)(ws + W_M2);
    _Float16* wc1  = (_Float16*)(ws + W_C1);
    _Float16* wc2  = (_Float16*)(ws + W_C2);
    float*    hbuf = (float*)   (ws + H_OFF);

    // region aliases (sequenced so lifetimes never overlap)
    float*    feats = (float*)   (ws + S2);
    _Float16* A0    = (_Float16*)(ws + S1);
    _Float16* nbuf  = (_Float16*)(ws + S2);
    _Float16* qkvh  = (_Float16*)(ws + S1);
    _Float16* obuf  = (_Float16*)(ws + S2);
    _Float16* n2buf = (_Float16*)(ws + S1);
    _Float16* gbuf  = (_Float16*)(ws + S2);
    _Float16* hf16  = (_Float16*)(ws + S1);
    _Float16* f1    = (_Float16*)(ws + S2);
    _Float16* f2    = (_Float16*)(ws + S2 + (16u << 20));

    const dim3 blk(256);
    auto cdiv = [](int a, int b) { return (a + b - 1) / b; };

    // ---- weight f32 -> f16 transposed staging ----
    peT_kernel<<<cdiv(HID * PDIMP, 256), blk, 0, stream>>>(pe_w, wpe);
    for (int i = 0; i < LAYERS; ++i) {
        castT_kernel<<<cdiv(HID*3*HID, 256), blk, 0, stream>>>(
            qkv_w + (size_t)i*HID*3*HID, wqkv + (size_t)i*HID*3*HID, HID, 3*HID);
        castT_kernel<<<cdiv(HID*HID, 256), blk, 0, stream>>>(
            out_w + (size_t)i*HID*HID, wout + (size_t)i*HID*HID, HID, HID);
        castT_kernel<<<cdiv(HID*MLPH, 256), blk, 0, stream>>>(
            mlp_w1 + (size_t)i*HID*MLPH, wm1 + (size_t)i*HID*MLPH, HID, MLPH);
        castT_kernel<<<cdiv(MLPH*HID, 256), blk, 0, stream>>>(
            mlp_w2 + (size_t)i*MLPH*HID, wm2 + (size_t)i*MLPH*HID, MLPH, HID);
    }
    castT_kernel<<<cdiv(CLSK*HID, 256), blk, 0, stream>>>(cls_w1, wc1, CLSK, HID);
    castT_kernel<<<cdiv(HID*128,  256), blk, 0, stream>>>(cls_w2, wc2, HID, 128);

    // ---- factors + normalize + pack ----
    factors_kernel<<<BATCH / 64, 64, 0, stream>>>(x, feats);
    normpack_kernel<<<cdiv(BATCH * TOTF, 256), blk, 0, stream>>>(feats, in_w, in_b, A0);

    // ---- patch embedding: h = A0 @ pe_w + pe_b (fp32 residual stream) ----
    gemm_f16<FBIAS | FF32><<<dim3(HID / 128, ROWS / 128), blk, 0, stream>>>(
        A0, wpe, pe_b, hbuf, nullptr, ROWS, HID, PDIMP);

    // ---- transformer layers ----
    for (int i = 0; i < LAYERS; ++i) {
        ln_kernel<<<ROWS / 8, blk, 0, stream>>>(hbuf, ln1_w + i*HID, ln1_b + i*HID, nbuf);
        gemm_f16<FBIAS | FF16><<<dim3(3*HID / 128, ROWS / 128), blk, 0, stream>>>(
            nbuf, wqkv + (size_t)i*HID*3*HID, qkv_b + i*3*HID,
            nullptr, qkvh, ROWS, 3*HID, HID);
        attn_kernel<<<BATCH, 96, 0, stream>>>(qkvh, obuf);
        gemm_f16<FBIAS | FRESID><<<dim3(HID / 128, ROWS / 128), blk, 0, stream>>>(
            obuf, wout + (size_t)i*HID*HID, out_b + i*HID,
            hbuf, nullptr, ROWS, HID, HID);
        ln_kernel<<<ROWS / 8, blk, 0, stream>>>(hbuf, ln2_w + i*HID, ln2_b + i*HID, n2buf);
        gemm_f16<FBIAS | FGELU | FF16><<<dim3(MLPH / 128, ROWS / 128), blk, 0, stream>>>(
            n2buf, wm1 + (size_t)i*HID*MLPH, mlp_b1 + i*MLPH,
            nullptr, gbuf, ROWS, MLPH, HID);
        gemm_f16<FBIAS | FRESID><<<dim3(HID / 128, ROWS / 128), blk, 0, stream>>>(
            gbuf, wm2 + (size_t)i*MLPH*HID, mlp_b2 + i*HID,
            hbuf, nullptr, ROWS, HID, MLPH);
    }

    // ---- classifier head ----
    cast_kernel<<<cdiv(ROWS * HID, 256), blk, 0, stream>>>(hbuf, hf16, ROWS * HID);
    gemm_f16<FBIAS | FGELU | FF16><<<dim3(512 / 128, BATCH / 128), blk, 0, stream>>>(
        hf16, wc1, cls_b1, nullptr, f1, BATCH, 512, CLSK);
    gemm_f16<FBIAS | FGELU | FF16><<<dim3(1, BATCH / 128), blk, 0, stream>>>(
        f1, wc2, cls_b2, nullptr, f2, BATCH, 128, 512);
    cls3_kernel<<<cdiv(BATCH, 256), blk, 0, stream>>>(f2, cls_w3, cls_b3, (float*)d_out);
}
